// AttnBlock_40072044871779
// MI455X (gfx1250) — matrix operand order repaired and checked by reading, not compiled, not run on hardware
//
#include <hip/hip_runtime.h>

// MI455X / gfx1250: wave32, WMMA bf16 16x16x32 (f32 accum) for all four GEMMs,
// double-buffered LDS staging with GLOBAL_LOAD_ASYNC_TO_LDS_B128 (ASYNCcnt) for
// the contiguous-copy tiles.

typedef __attribute__((ext_vector_type(16))) __bf16        v16bf;
typedef __attribute__((ext_vector_type(8)))  float         v8f;
typedef __attribute__((ext_vector_type(8)))  unsigned int  v8u;
typedef __attribute__((ext_vector_type(4)))  unsigned int  v4u;

#define BB   8
#define CC   512
#define NN   4096
#define GG   8
#define CPG  64          // C / G

__device__ __forceinline__ unsigned short f2bf(float f) {
    unsigned u = __builtin_bit_cast(unsigned, f);
    unsigned r = u + 0x7FFFu + ((u >> 16) & 1u);   // round-to-nearest-even
    return (unsigned short)(r >> 16);
}
__device__ __forceinline__ float bf2f(unsigned short h) {
    return __builtin_bit_cast(float, (unsigned)h << 16);
}

// ---- CDNA5 async global->LDS copy (ISA §15.18 op 98, tracked by ASYNCcnt) ----
__device__ __forceinline__ unsigned lds_off(const void* p) {
    // flat shared address: low 32 bits are the LDS byte offset (aperture rule)
    return (unsigned)(size_t)p;
}
__device__ __forceinline__ void async_ld_b128(unsigned lds_byte_off, const void* g) {
    asm volatile("global_load_async_to_lds_b128 %0, %1, off"
                 :: "v"(lds_byte_off), "v"(g) : "memory");
}
__device__ __forceinline__ void wait_async0() {
    asm volatile("s_wait_asynccnt 0x0" ::: "memory");
}

// ---------------------------------------------------------------- weights -> bf16
__global__ void convert_weights(const float* __restrict__ qkv_w,
                                const float* __restrict__ proj_w,
                                unsigned short* __restrict__ wq,
                                unsigned short* __restrict__ wp) {
    int i = blockIdx.x * 256 + threadIdx.x;
    const int t1 = 3 * CC * CC;
    if (i < t1) wq[i] = f2bf(qkv_w[i]);
    int j = i - t1;
    if (j >= 0 && j < CC * CC) wp[j] = f2bf(proj_w[j]);
}

// ---------------------------------------------------------------- GroupNorm stats
__global__ void gn_stats(const float* __restrict__ x, float* __restrict__ stats) {
    __shared__ float s1[256], s2[256];
    const int bg = blockIdx.x;                       // b*G + g (group chunk is contiguous)
    const float* base = x + (size_t)bg * CPG * NN;
    float sum = 0.f, ss = 0.f;
    for (int i = threadIdx.x; i < CPG * NN; i += 256) {
        float v = base[i];
        sum += v; ss += v * v;
    }
    s1[threadIdx.x] = sum; s2[threadIdx.x] = ss;
    __syncthreads();
    for (int o = 128; o > 0; o >>= 1) {
        if (threadIdx.x < o) {
            s1[threadIdx.x] += s1[threadIdx.x + o];
            s2[threadIdx.x] += s2[threadIdx.x + o];
        }
        __syncthreads();
    }
    if (threadIdx.x == 0) {
        const float inv = 1.0f / (float)(CPG * NN);
        float mean = s1[0] * inv;
        float var  = s2[0] * inv - mean * mean;
        stats[bg * 2 + 0] = mean;
        stats[bg * 2 + 1] = rsqrtf(var + 1e-6f);
    }
}

// ---------------------------------------------------------------- normalize -> bf16
__global__ void gn_apply(const float* __restrict__ x, const float* __restrict__ stats,
                         const float* __restrict__ scale, const float* __restrict__ bias,
                         unsigned short* __restrict__ hbf) {
    size_t i = (size_t)blockIdx.x * 256 + threadIdx.x;     // B*C*N = 2^24 elements
    int c  = (int)((i >> 12) & (CC - 1));                  // N = 2^12
    int bg = (int)(i >> 18);                               // CPG*N = 2^18
    float mean = stats[bg * 2 + 0], rstd = stats[bg * 2 + 1];
    hbf[i] = f2bf((x[i] - mean) * rstd * scale[c] + bias[c]);
}

// ---------------------------------------------------------------- softmax over tokens (k rows)
__global__ void softmax_rows(unsigned short* __restrict__ qkv_bf) {
    const int row = blockIdx.x;                 // b*C + d
    const int b = row >> 9, d = row & (CC - 1);
    unsigned short* p = qkv_bf + ((size_t)b * 3 * CC + CC + d) * NN;
    __shared__ float red[256];
    float m = -1e30f;
    for (int i = threadIdx.x; i < NN; i += 256) m = fmaxf(m, bf2f(p[i]));
    red[threadIdx.x] = m; __syncthreads();
    for (int o = 128; o > 0; o >>= 1) {
        if (threadIdx.x < o) red[threadIdx.x] = fmaxf(red[threadIdx.x], red[threadIdx.x + o]);
        __syncthreads();
    }
    const float mx = red[0]; __syncthreads();
    float s = 0.f;
    for (int i = threadIdx.x; i < NN; i += 256) s += __expf(bf2f(p[i]) - mx);
    red[threadIdx.x] = s; __syncthreads();
    for (int o = 128; o > 0; o >>= 1) {
        if (threadIdx.x < o) red[threadIdx.x] += red[threadIdx.x + o];
        __syncthreads();
    }
    const float inv = 1.0f / red[0];
    for (int i = threadIdx.x; i < NN; i += 256) p[i] = f2bf(__expf(bf2f(p[i]) - mx) * inv);
}

// ---------------------------------------------------------------- WMMA GEMM
// D = A[M,K] x B[K,N] (+bias) per batch. 256 threads = 8 waves; block tile 128x64,
// each wave owns a 16x64 strip (4 accumulator fragments). K-step 32, double-buffered.
// A tile (and Bt tile when TRANSB) staged via global_load_async_to_lds_b128.
// MODE 0: store bf16 natural [M,N], +bias, rows < scaleRows scaled by scaleVal
// MODE 1: store bf16 transposed [N,M] (packed 16B stores) — used for context^T
// MODE 2: store f32 natural [M,N], +bias, +X residual
// TRANSB: B operand given as Bt[N,K] row-major (reads column-major B).
template<int MODE, bool TRANSB>
__global__ __launch_bounds__(256) void gemm_wmma(
        const unsigned short* __restrict__ A,  long long Abs,
        const unsigned short* __restrict__ Bm, long long Bbs,
        void* __restrict__ Dv,                 long long Dbs,
        const float* __restrict__ bias,
        const float* __restrict__ X,           long long Xbs,
        int M, int Nn, int K, int scaleRows, float scaleVal)
{
    __shared__ __align__(16) unsigned int   sA[2][128 * 16];  // [buf][row][k/2]  16 KB
    __shared__ __align__(16) unsigned short sB[2][64 * 32];   // [buf][n][k]       8 KB
    const int tid = threadIdx.x, wave = tid >> 5, lane = tid & 31;
    const int lidx = lane & 15, lhalf = lane >> 4;
    const int b  = blockIdx.z;
    const int n0 = blockIdx.x * 64, m0 = blockIdx.y * 128;
    const unsigned int*   gA = (const unsigned int*)(A + (size_t)b * Abs);
    const unsigned short* gB = Bm + (size_t)b * Bbs;
    const int Ku = K >> 1;

    v8f acc[4] = {};

    // ---- staging helpers -------------------------------------------------
    auto stageA = [&](int k0, int bufi) {        // 128 rows x 64 B -> 2 async b128/thread
        const int kuo = k0 >> 1;
        #pragma unroll
        for (int j = 0; j < 2; j++) {
            const int c = tid + j * 256;         // 0..511 chunks of 16 B
            const int row = c >> 2, qu = (c & 3) << 2;
            async_ld_b128(lds_off(&sA[bufi][row * 16 + qu]),
                          gA + (size_t)(m0 + row) * Ku + kuo + qu);
        }
    };
    auto stageBt = [&](int k0, int bufi) {       // 64 rows x 64 B -> 1 async b128/thread
        const int kuo = k0 >> 1;
        const int row = tid >> 2, qu = (tid & 3) << 2;
        async_ld_b128(lds_off((unsigned int*)sB[bufi] + row * 16 + qu),
                      (const unsigned int*)gB + (size_t)(n0 + row) * Ku + kuo + qu);
    };
    auto loadBrow = [&](int k0) -> v4u {         // B[K,N]: 8 halves of one k-row
        const int kk = tid >> 3, nb = (tid & 7) * 8;
        return *(const v4u*)(gB + (size_t)(k0 + kk) * Nn + n0 + nb);
    };
    auto storeBrow = [&](v4u r, int bufi) {      // transpose into LDS [n][k]
        const int kk = tid >> 3, nb = (tid & 7) * 8;
        #pragma unroll
        for (int i = 0; i < 4; i++) {
            sB[bufi][(nb + 2 * i    ) * 32 + kk] = (unsigned short)(r[i] & 0xFFFFu);
            sB[bufi][(nb + 2 * i + 1) * 32 + kk] = (unsigned short)(r[i] >> 16);
        }
    };
    auto compute = [&](int bufi) {
        // A fragment: 16x32 bf16 (ISA: K pairs at (v&3)*2 + (v>>2)*16 + half*8)
        v8u au;
        const int arow = wave * 16 + lidx;
        #pragma unroll
        for (int v = 0; v < 8; v++) {
            const int k = ((v & 3) << 1) + ((v >> 2) << 4) + (lhalf << 3);
            au[v] = sA[bufi][arow * 16 + (k >> 1)];
        }
        const v16bf af = __builtin_bit_cast(v16bf, au);
        const unsigned int* sBu = (const unsigned int*)sB[bufi];
        #pragma unroll
        for (int ns = 0; ns < 4; ns++) {
            // B fragment: 32x16 bf16 (K pairs at (v&3)*2 + (v>>2)*16 + half*8, column = lane&15)
            v8u bu;
            #pragma unroll
            for (int v = 0; v < 8; v++)
                bu[v] = sBu[(ns * 16 + lidx) * 16 + (lhalf << 2) + (v & 3) + ((v >> 2) << 3)];
            const v16bf bfr = __builtin_bit_cast(v16bf, bu);
            acc[ns] = __builtin_amdgcn_wmma_f32_16x16x32_bf16(
                          false, af, false, bfr, (short)0, acc[ns], false, false);
        }
    };

    // ---- pipelined main loop --------------------------------------------
    const int nt = K >> 5;
    stageA(0, 0);
    if (TRANSB) stageBt(0, 0);
    else        storeBrow(loadBrow(0), 0);
    wait_async0();
    __syncthreads();

    for (int t = 0; t < nt; t++) {
        const bool more = (t + 1 < nt);
        v4u bnext = {};
        if (more) {
            stageA((t + 1) << 5, (t + 1) & 1);
            if (TRANSB) stageBt((t + 1) << 5, (t + 1) & 1);
            else        bnext = loadBrow((t + 1) << 5);
        }
        compute(t & 1);
        if (more) {
            if (!TRANSB) storeBrow(bnext, (t + 1) & 1);
            wait_async0();
            __syncthreads();
        }
    }

    // ---- epilogue. C/D layout: n = lane&15, m = i + 8*(lane>>4). ---------
    if (MODE == 0) {
        unsigned short* D = (unsigned short*)Dv + (size_t)b * Dbs;
        #pragma unroll
        for (int ns = 0; ns < 4; ns++) {
            const int col = n0 + ns * 16 + lidx;
            #pragma unroll
            for (int i = 0; i < 8; i++) {
                const int m = m0 + wave * 16 + (lhalf << 3) + i;
                float v = acc[ns][i];
                if (bias) v += bias[m];
                if (m < scaleRows) v *= scaleVal;
                D[(size_t)m * Nn + col] = f2bf(v);
            }
        }
    } else if (MODE == 1) {
        unsigned short* Dt = (unsigned short*)Dv + (size_t)b * Dbs;   // [Nn][M]
        const int dbase = m0 + wave * 16 + (lhalf << 3);
        #pragma unroll
        for (int ns = 0; ns < 4; ns++) {
            const int e = n0 + ns * 16 + lidx;
            v4u p;
            #pragma unroll
            for (int j = 0; j < 4; j++)
                p[j] = (unsigned)f2bf(acc[ns][2 * j]) |
                       ((unsigned)f2bf(acc[ns][2 * j + 1]) << 16);
            *(v4u*)(Dt + (size_t)e * M + dbase) = p;   // 16B aligned: dbase % 8 == 0
        }
    } else {
        float* D = (float*)Dv + (size_t)b * Dbs;
        const float* Xb = X + (size_t)b * Xbs;
        #pragma unroll
        for (int ns = 0; ns < 4; ns++) {
            const int col = n0 + ns * 16 + lidx;
            #pragma unroll
            for (int i = 0; i < 8; i++) {
                const int m = m0 + wave * 16 + (lhalf << 3) + i;
                const size_t off = (size_t)m * Nn + col;
                D[off] = acc[ns][i] + (bias ? bias[m] : 0.f) + Xb[off];
            }
        }
    }
}

// ---------------------------------------------------------------- launch
extern "C" void kernel_launch(void* const* d_in, const int* in_sizes, int n_in,
                              void* d_out, int out_size, void* d_ws, size_t ws_size,
                              hipStream_t stream) {
    const float* x      = (const float*)d_in[0];
    const float* qkv_w  = (const float*)d_in[1];
    const float* qkv_b  = (const float*)d_in[2];
    const float* proj_w = (const float*)d_in[3];
    const float* proj_b = (const float*)d_in[4];
    const float* gn_s   = (const float*)d_in[5];
    const float* gn_b   = (const float*)d_in[6];
    float* out = (float*)d_out;

    char* ws = (char*)d_ws;
    unsigned short* h_bf   = (unsigned short*)(ws);               // B*C*N        bf16  32 MB
    unsigned short* qkv_bf = (unsigned short*)(ws +  33554432);   // B*3C*N       bf16  96 MB
    unsigned short* ctx_bf = (unsigned short*)(ws + 134217728);   // B*C*C (ctx^T)bf16   4 MB
    unsigned short* out_bf = (unsigned short*)(ws + 138412032);   // B*C*N        bf16  32 MB
    unsigned short* wq_bf  = (unsigned short*)(ws + 171966464);   // 3C*C         bf16
    unsigned short* wp_bf  = (unsigned short*)(ws + 173539328);   // C*C          bf16
    float*          stats  = (float*)(ws + 174063616);            // B*G*{mean,rstd}

    convert_weights<<<4096, 256, 0, stream>>>(qkv_w, proj_w, wq_bf, wp_bf);
    gn_stats<<<BB * GG, 256, 0, stream>>>(x, stats);
    gn_apply<<<65536, 256, 0, stream>>>(x, stats, gn_s, gn_b, h_bf);

    const float rs = 0.04419417382415922f;   // 512^-0.5

    // qkv = Wq @ h  (M=1536, N=4096, K=512); q rows scaled, bias added, bf16 out
    gemm_wmma<0, false><<<dim3(64, 12, BB), 256, 0, stream>>>(
        wq_bf, 0, h_bf, (long long)CC * NN, qkv_bf, (long long)3 * CC * NN,
        qkv_b, nullptr, 0, 3 * CC, NN, CC, CC, rs);

    softmax_rows<<<BB * CC, 256, 0, stream>>>(qkv_bf);

    // context^T[e][d] = sum_n k[d,n] v[e,n]  (M=d=512, N=e=512, K=n=4096, B from v^T)
    gemm_wmma<1, true><<<dim3(8, 4, BB), 256, 0, stream>>>(
        qkv_bf + (size_t)CC * NN,     (long long)3 * CC * NN,   // k region
        qkv_bf + (size_t)2 * CC * NN, (long long)3 * CC * NN,   // v region as Bt[N,K]
        ctx_bf, (long long)CC * CC,
        nullptr, nullptr, 0, CC, CC, NN, 0, 1.0f);

    // out[e,n] = sum_d ctx^T[e,d] q[d,n]  (M=512, N=4096, K=512)
    gemm_wmma<0, false><<<dim3(64, 4, BB), 256, 0, stream>>>(
        ctx_bf, (long long)CC * CC, qkv_bf, (long long)3 * CC * NN,
        out_bf, (long long)CC * NN, nullptr, nullptr, 0, CC, NN, CC, 0, 1.0f);

    // y = Wp @ out + proj_b + x   (fp32 residual output)
    gemm_wmma<2, false><<<dim3(64, 4, BB), 256, 0, stream>>>(
        wp_bf, 0, out_bf, (long long)CC * NN, out, (long long)CC * NN,
        proj_b, x, (long long)CC * NN, CC, NN, CC, 0, 1.0f);

    (void)in_sizes; (void)n_in; (void)out_size; (void)ws_size;
}
